// ParticleFlowNetwork_88502096101647
// MI455X (gfx1250) — matrix operand order repaired and checked by reading, not compiled, not run on hardware
//
#include <hip/hip_runtime.h>

// ---------------------------------------------------------------------------
// MI455X (gfx1250) fused ParticleFlowNetwork forward.
//
// Roofline: node MLP = 35 GFLOP vs ~10MB HBM traffic -> compute bound >10x at
// 23.3 TB/s.  v_wmma_f32_16x16x32_bf16 (bf16 in, f32 acc).  Weights are
// staged per-block into LDS with global_load_async_to_lds_b128 (ASYNCcnt),
// overlapped with the VALU layer-0, cutting B-fragment L2 reads 8x.  The WMMA
// inner loops are double-buffered (prefetch distance 2) so the compiler can
// wait on dscnt<=2 instead of draining DS before every WMMA.
// ---------------------------------------------------------------------------

typedef __attribute__((ext_vector_type(16))) __bf16       v16bf;
typedef __attribute__((ext_vector_type(8)))  float        v8f;
typedef __attribute__((ext_vector_type(4))) unsigned int  u32x4;
typedef unsigned short u16;

union Frag32 {           // one 16x32 bf16 WMMA operand fragment per lane: 32B
  u32x4 q[2];
  v16bf bf;
};

__device__ __forceinline__ u16 f2bf(float f) {             // RNE f32 -> bf16
  unsigned int u = __builtin_bit_cast(unsigned int, f);
  u += 0x7FFFu + ((u >> 16) & 1u);
  return (u16)(u >> 16);
}
__device__ __forceinline__ float bf2f(u16 h) {
  unsigned int u = ((unsigned int)h) << 16;
  return __builtin_bit_cast(float, u);
}

// gfx1250 async global->LDS copy, GV mode (64-bit vaddr, no saddr).
// LDS byte address taken from the lds_off VGPR; tracked with ASYNCcnt.
__device__ __forceinline__ void async_ld_b128(unsigned lds_off, const void* gptr) {
  asm volatile("global_load_async_to_lds_b128 %0, %1, off"
               :: "v"(lds_off), "v"(gptr)
               : "memory");
}
__device__ __forceinline__ void wait_async0() {
  asm volatile("s_wait_asynccnt 0x0" ::: "memory");
}
// Generic LDS pointer -> raw LDS byte offset (flat LDS aperture keeps the LDS
// offset in addr[31:0], ISA 10.2 aperture mapping).
__device__ __forceinline__ unsigned lds_off_of(const void* p) {
  return (unsigned)(unsigned long long)p;
}

// Load one swizzled B fragment (two 16B chunks) from an LDS weight bank.
__device__ __forceinline__ void ld_wfrag(Frag32& f, const u32x4* wv, int tile,
                                         int wchunk) {
  const int bi = tile * 68 + wchunk;
  f.q[0] = wv[bi];
  f.q[1] = wv[bi + 1];
}
// Load one dense B fragment straight from a packed global buffer.
__device__ __forceinline__ void ld_gfrag(Frag32& f, const u32x4* wq, int tile,
                                         int lane) {
  const int bi = (tile * 32 + lane) * 2;
  f.q[0] = wq[bi];
  f.q[1] = wq[bi + 1];
}

// ---------------------------------------------------------------------------
// Pack a row-major f32 weight matrix W[K][N] into WMMA B-fragment order, bf16.
// B (32x16, 16-bit) layout (ISA 7.12.2): lane = N within tile; lanes 0-15 hold
// K=kc*32+0..7,16..23 (2 halves / VGPR), lanes 16-31 hold K=kc*32+8..15,24..31.
// Buffer index: (((kc*nTiles + nt)*32 + lane)*16 + q)  -> 32B contig per lane.
// ---------------------------------------------------------------------------
__global__ void pack_b_kernel(const float* __restrict__ W, u16* __restrict__ out,
                              int K, int N) {
  const int nT = N >> 4;
  const int total = (K >> 5) * nT * 512;
  for (int i = blockIdx.x * blockDim.x + threadIdx.x; i < total;
       i += gridDim.x * blockDim.x) {
    const int q    = i & 15;
    const int lane = (i >> 4) & 31;
    const int ntk  = i >> 9;
    const int nt   = ntk % nT;
    const int kc   = ntk / nT;
    const int k = kc * 32 + (lane >> 4) * 8 + (q & 7) + (q >> 3) * 16;
    const int n = nt * 16 + (lane & 15);
    out[i] = f2bf(W[k * N + n]);
  }
}

__global__ void zero_kernel(float* __restrict__ p, int n) {
  const int i = blockIdx.x * blockDim.x + threadIdx.x;
  if (i < n) p[i] = 0.0f;
}

// ---------------------------------------------------------------------------
// Fused per-node phi MLP + segment-sum.  Block = 256 thr (8 waves), 128 nodes.
//
// Dynamic LDS layout (halves):
//   [0,17408)              H tile, 128 rows x 136 (272B row: 16B-aligned uint4
//                          loads, 4-bank rotation per row)
//   [17408, 34816)         W1 fragments, 32 tiles x 544 halves (1088B/tile)
//   [34816, 52224)         W2 fragments
// Weight tile layout: lane's 32B at  lane*32 + (lane>>3)*16  (+16B pad per 8
// lanes) -> the 32 per-lane ds_load_b128 of a B fragment hit disjoint banks.
// ---------------------------------------------------------------------------
__global__ __launch_bounds__(256) void phi_nodes_kernel(
    const float* __restrict__ x, const int* __restrict__ batch,
    const float* __restrict__ w0, const float* __restrict__ b0,
    const float* __restrict__ b1, const float* __restrict__ b2,
    const u16* __restrict__ w1p, const u16* __restrict__ w2p,
    float* __restrict__ g) {
  extern __shared__ u16 smem[];
  u16* lds_h  = smem;                 // 128*136
  u16* lds_w1 = smem + 128 * 136;     // 32 tiles * 544
  u16* lds_w2 = lds_w1 + 32 * 544;
  __shared__ float lds_w0[4 * 128];
  __shared__ float lds_x[128 * 4];
  __shared__ float lds_b0[128];

  const int t = threadIdx.x;
  const int nodeBase = blockIdx.x * 128;
  (void)w2p;  // contiguous after w1p in workspace

  // ---- kick off async weight staging (overlaps with layer 0 below) --------
  {
    const char* src = (const char*)w1p;       // w2p follows contiguously
    const unsigned dst0 = lds_off_of(lds_w1);
    for (int j = t; j < 4096; j += 256) {     // 4096 x 16B = W1+W2
      const int jj   = j & 2047;
      const int tile = jj >> 6;
      const int sc   = jj & 63;
      const unsigned dst = dst0 + (unsigned)((j >> 11) * 34816 + tile * 1088 +
                                             (sc >> 4) * 272 + (sc & 15) * 16);
      async_ld_b128(dst, src + (size_t)j * 16);
    }
  }

  if (t < 128) lds_b0[t] = b0[t];
  for (int i = t; i < 512; i += 256) {
    lds_w0[i] = w0[i];
    lds_x[i]  = x[nodeBase * 4 + i];
  }
  __syncthreads();

  // ---- layer 0: [128,4]@[4,128], K=4 -> plain FMAs, relu, bf16 into LDS ----
  for (int e = t; e < 128 * 128; e += 256) {
    const int r = e >> 7, c = e & 127;
    float s = lds_b0[c];
#pragma unroll
    for (int k = 0; k < 4; ++k) s = fmaf(lds_x[r * 4 + k], lds_w0[k * 128 + c], s);
    lds_h[r * 136 + c] = f2bf(fmaxf(s, 0.0f));
  }
  wait_async0();          // own async staging complete
  __syncthreads();        // everyone's staging + layer-0 tile visible

  const int lane = t & 31;
  const int wave = t >> 5;
  const int rowBase = wave * 16;         // 16 nodes per wave
  const int col   = lane & 15;
  const int lhalf = lane >> 4;           // A layout: lanes 16-31 carry K+8
  const u32x4* lsv = (const u32x4*)lds_h;
  const int wchunk = lane * 2 + (lane >> 3);   // swizzled fragment chunk base

  // ---- layer 1: WMMA, M=16 N=128(8 tiles) K=128(4 chunks) ----
  Frag32 a[4];
#pragma unroll
  for (int kc = 0; kc < 4; ++kc) {       // A 16-bit 16x32 layout (ISA 7.12.2)
    const int idx = (rowBase + col) * 17 + kc * 4 + lhalf;   // 17 uint4 / row
    a[kc].q[0] = lsv[idx];               // K = base+0..7
    a[kc].q[1] = lsv[idx + 2];           // K = base+16..23
  }

  v8f acc[8];
#pragma unroll
  for (int nt = 0; nt < 8; ++nt) {       // hoisted bias splat (C layout: N=lane&15)
    const float bv = b1[nt * 16 + col];
#pragma unroll
    for (int e = 0; e < 8; ++e) acc[nt][e] = bv;
  }
  {
    const u32x4* w1v = (const u32x4*)lds_w1;
    Frag32 bw[2];                        // double buffer, prefetch distance 2
    ld_wfrag(bw[0], w1v, 0, wchunk);     // i=0: kc0 nt0 -> tile 0
    ld_wfrag(bw[1], w1v, 8, wchunk);     // i=1: kc1 nt0 -> tile 8
#pragma unroll
    for (int i = 0; i < 32; ++i) {       // i = nt*4 + kc
      const int nt = i >> 2, kc = i & 3;
      acc[nt] = __builtin_amdgcn_wmma_f32_16x16x32_bf16(
          false, a[kc].bf, false, bw[i & 1].bf, (short)0, acc[nt], false, false);
      if (i + 2 < 32) {
        const int j = i + 2;
        ld_wfrag(bw[i & 1], w1v, (j & 3) * 8 + (j >> 2), wchunk);
      }
    }
  }
  __syncthreads();
  // relu + bf16 back into LDS (C/D layout: VGPR e -> row e / 8+e)
#pragma unroll
  for (int nt = 0; nt < 8; ++nt)
#pragma unroll
    for (int e = 0; e < 8; ++e) {
      const int r = rowBase + lhalf * 8 + e;
      lds_h[r * 136 + nt * 16 + col] = f2bf(fmaxf(acc[nt][e], 0.0f));
    }
  __syncthreads();

  // ---- layer 2: same shape, no relu ----
#pragma unroll
  for (int kc = 0; kc < 4; ++kc) {
    const int idx = (rowBase + col) * 17 + kc * 4 + lhalf;
    a[kc].q[0] = lsv[idx];
    a[kc].q[1] = lsv[idx + 2];
  }
#pragma unroll
  for (int nt = 0; nt < 8; ++nt) {
    const float bv = b2[nt * 16 + col];
#pragma unroll
    for (int e = 0; e < 8; ++e) acc[nt][e] = bv;
  }
  {
    const u32x4* w2v = (const u32x4*)lds_w2;
    Frag32 bw[2];
    ld_wfrag(bw[0], w2v, 0, wchunk);
    ld_wfrag(bw[1], w2v, 8, wchunk);
#pragma unroll
    for (int i = 0; i < 32; ++i) {
      const int nt = i >> 2, kc = i & 3;
      acc[nt] = __builtin_amdgcn_wmma_f32_16x16x32_bf16(
          false, a[kc].bf, false, bw[i & 1].bf, (short)0, acc[nt], false, false);
      if (i + 2 < 32) {
        const int j = i + 2;
        ld_wfrag(bw[i & 1], w2v, (j & 3) * 8 + (j >> 2), wchunk);
      }
    }
  }

  // ---- segment-sum: batch sorted -> compress 8-row runs before atomics ----
  int ids[8];
  const int nodeRow = nodeBase + rowBase + lhalf * 8;
#pragma unroll
  for (int e = 0; e < 8; ++e) ids[e] = batch[nodeRow + e];
#pragma unroll
  for (int nt = 0; nt < 8; ++nt) {
    const int c = nt * 16 + col;
    float run = acc[nt][0];
    int   cur = ids[0];
#pragma unroll
    for (int e = 1; e < 8; ++e) {
      if (ids[e] == cur) {
        run += acc[nt][e];
      } else {
        atomicAdd(&g[cur * 128 + c], run);
        cur = ids[e];
        run = acc[nt][e];
      }
    }
    atomicAdd(&g[cur * 128 + c], run);
  }
}

// ---------------------------------------------------------------------------
// Head MLP on pooled graphs: [4096,128]->256->256->2.  Block = 128 thr
// (4 waves), 64 graphs/block.  Tiny (0.8 GFLOP, ~49MB L2) -> B frags straight
// from L2, double-buffered.  LDS tile stride 264 halves (528B, 33 uint4).
// ---------------------------------------------------------------------------
__global__ __launch_bounds__(128) void head_kernel(
    const float* __restrict__ g,
    const u16* __restrict__ fw0p, const float* __restrict__ fb0,
    const u16* __restrict__ fw1p, const float* __restrict__ fb1,
    const float* __restrict__ fw2, const float* __restrict__ fb2,
    float* __restrict__ out) {
  __shared__ u16 lds[64 * 264];
  const int t = threadIdx.x;
  const int gBase = blockIdx.x * 64;

  for (int e = t; e < 64 * 128; e += 128) {
    const int r = e >> 7, c = e & 127;
    lds[r * 264 + c] = f2bf(g[(gBase + r) * 128 + c]);
  }
  __syncthreads();

  const int lane = t & 31, wave = t >> 5;
  const int rowBase = wave * 16, col = lane & 15, lhalf = lane >> 4;
  const u32x4* lsv = (const u32x4*)lds;

  // ---- f0: K=128 (4 chunks), N=256 (16 tiles, 2 groups to bound VGPRs) ----
  Frag32 a4[4];
#pragma unroll
  for (int kc = 0; kc < 4; ++kc) {
    const int idx = (rowBase + col) * 33 + kc * 4 + lhalf;
    a4[kc].q[0] = lsv[idx];
    a4[kc].q[1] = lsv[idx + 2];
  }
  __syncthreads();
  const u32x4* w0q = (const u32x4*)fw0p;
  for (int grp = 0; grp < 2; ++grp) {
    v8f acc[8];
#pragma unroll
    for (int nt = 0; nt < 8; ++nt) {
      const float bv = fb0[(grp * 8 + nt) * 16 + col];
#pragma unroll
      for (int e = 0; e < 8; ++e) acc[nt][e] = bv;
    }
    Frag32 bw[2];                        // tile index = kc*16 + grp*8 + nt
    ld_gfrag(bw[0], w0q, grp * 8, lane);           // i=0: kc0 nt0
    ld_gfrag(bw[1], w0q, 16 + grp * 8, lane);      // i=1: kc1 nt0
#pragma unroll
    for (int i = 0; i < 32; ++i) {       // i = nt*4 + kc
      const int nt = i >> 2, kc = i & 3;
      acc[nt] = __builtin_amdgcn_wmma_f32_16x16x32_bf16(
          false, a4[kc].bf, false, bw[i & 1].bf, (short)0, acc[nt], false, false);
      if (i + 2 < 32) {
        const int j = i + 2;
        ld_gfrag(bw[i & 1], w0q, (j & 3) * 16 + grp * 8 + (j >> 2), lane);
      }
    }
#pragma unroll
    for (int nt = 0; nt < 8; ++nt)
#pragma unroll
      for (int e = 0; e < 8; ++e) {
        const int r = rowBase + lhalf * 8 + e;
        lds[r * 264 + grp * 128 + nt * 16 + col] = f2bf(fmaxf(acc[nt][e], 0.0f));
      }
  }
  __syncthreads();

  // ---- f1: K=256 (8 chunks), N=256 ----
  Frag32 a8[8];
#pragma unroll
  for (int kc = 0; kc < 8; ++kc) {
    const int idx = (rowBase + col) * 33 + kc * 4 + lhalf;
    a8[kc].q[0] = lsv[idx];
    a8[kc].q[1] = lsv[idx + 2];
  }
  __syncthreads();
  const u32x4* w1q = (const u32x4*)fw1p;
  for (int grp = 0; grp < 2; ++grp) {
    v8f acc[8];
#pragma unroll
    for (int nt = 0; nt < 8; ++nt) {
      const float bv = fb1[(grp * 8 + nt) * 16 + col];
#pragma unroll
      for (int e = 0; e < 8; ++e) acc[nt][e] = bv;
    }
    Frag32 bw[2];                        // tile index = kc*16 + grp*8 + nt
    ld_gfrag(bw[0], w1q, grp * 8, lane);           // i=0: kc0 nt0
    ld_gfrag(bw[1], w1q, 16 + grp * 8, lane);      // i=1: kc1 nt0
#pragma unroll
    for (int i = 0; i < 64; ++i) {       // i = nt*8 + kc
      const int nt = i >> 3, kc = i & 7;
      acc[nt] = __builtin_amdgcn_wmma_f32_16x16x32_bf16(
          false, a8[kc].bf, false, bw[i & 1].bf, (short)0, acc[nt], false, false);
      if (i + 2 < 64) {
        const int j = i + 2;
        ld_gfrag(bw[i & 1], w1q, (j & 7) * 16 + grp * 8 + (j >> 3), lane);
      }
    }
#pragma unroll
    for (int nt = 0; nt < 8; ++nt)
#pragma unroll
      for (int e = 0; e < 8; ++e) {
        const int r = rowBase + lhalf * 8 + e;
        lds[r * 264 + grp * 128 + nt * 16 + col] = f2bf(fmaxf(acc[nt][e], 0.0f));
      }
  }
  __syncthreads();

  // ---- f2: [64,256] @ [256,2] + b, one output element per thread ----
  {
    const int r = t >> 1, c = t & 1;
    float s = fb2[c];
    for (int k = 0; k < 256; ++k)
      s = fmaf(bf2f(lds[r * 264 + k]), fw2[k * 2 + c], s);
    out[(gBase + r) * 2 + c] = s;
  }
}

// ---------------------------------------------------------------------------
extern "C" void kernel_launch(void* const* d_in, const int* in_sizes, int n_in,
                              void* d_out, int out_size, void* d_ws, size_t ws_size,
                              hipStream_t stream) {
  (void)in_sizes; (void)n_in; (void)out_size; (void)ws_size;
  const float* x      = (const float*)d_in[0];
  // d_in[1] edge_index: mathematically dead (update ignores aggr_out)
  const int*   batch  = (const int*)d_in[2];
  const float* phi_w0 = (const float*)d_in[3];
  const float* phi_b0 = (const float*)d_in[4];
  const float* phi_w1 = (const float*)d_in[5];
  const float* phi_b1 = (const float*)d_in[6];
  const float* phi_w2 = (const float*)d_in[7];
  const float* phi_b2 = (const float*)d_in[8];
  const float* f_w0   = (const float*)d_in[9];
  const float* f_b0   = (const float*)d_in[10];
  const float* f_w1   = (const float*)d_in[11];
  const float* f_b1   = (const float*)d_in[12];
  const float* f_w2   = (const float*)d_in[13];
  const float* f_b2   = (const float*)d_in[14];
  float* out = (float*)d_out;

  // workspace layout: pooled g (2MB) + packed bf16 weight fragments (256KB)
  float* g   = (float*)d_ws;
  u16* w1p   = (u16*)((char*)d_ws + (size_t)(4096 * 128) * sizeof(float));
  u16* w2p   = w1p  + 128 * 128;    // must stay contiguous after w1p
  u16* fw0p  = w2p  + 128 * 128;
  u16* fw1p  = fw0p + 128 * 256;

  zero_kernel<<<(4096 * 128 + 255) / 256, 256, 0, stream>>>(g, 4096 * 128);
  pack_b_kernel<<<64,  256, 0, stream>>>(phi_w1, w1p,  128, 128);
  pack_b_kernel<<<64,  256, 0, stream>>>(phi_w2, w2p,  128, 128);
  pack_b_kernel<<<128, 256, 0, stream>>>(f_w0,   fw0p, 128, 256);
  pack_b_kernel<<<256, 256, 0, stream>>>(f_w1,   fw1p, 256, 256);

  // dynamic LDS: H tile (34,816B) + swizzled W1 + W2 fragments (2 x 34,816B)
  const size_t phi_lds = (size_t)(128 * 136 + 2 * 32 * 544) * sizeof(u16);
  phi_nodes_kernel<<<4096, 256, phi_lds, stream>>>(x, batch, phi_w0, phi_b0,
                                                   phi_b1, phi_b2, w1p, w2p, g);
  head_kernel<<<64, 128, 0, stream>>>(g, fw0p, f_b0, fw1p, f_b1, f_w2, f_b2, out);
}
